// LFMMILoss_15848429322334
// MI455X (gfx1250) — compile-verified, matrix operand order checked by reading, not run on hardware
//
#include <hip/hip_runtime.h>

typedef __attribute__((ext_vector_type(16))) _Float16 v16h;
typedef __attribute__((ext_vector_type(8)))  float    v8f;

#define BB    32
#define TLEN  500
#define PP    2048
#define SDEN  1024
#define SNUM  128

// ------------------------------------------------------------------
// ws layout: [0..1] float accumulators (num_llh, den_llh),
//            +256 bytes: packed f16 B-matrix Tb[nTile][kc][lane][half]
//            (1024*1024 halfs = 2 MB), laid out exactly as WMMA f16
//            16x16x32 B-fragments so each lane loads 32 contiguous bytes.
// Fragment element map (B is KxN=32x16): lane L, half h ->
//   N = L%16 ;  K = 16*(h/8) + 8*(L/16) + (h%8)
// ------------------------------------------------------------------

__global__ __launch_bounds__(512)
void prep_kernel(const float* __restrict__ den_logT,
                 _Float16* __restrict__ Tb, float* __restrict__ acc) {
  int idx = blockIdx.x * blockDim.x + threadIdx.x;
  if (idx < 2) acc[idx] = 0.0f;
  if (idx < SDEN * SDEN) {
    int h  = idx & 15;
    int L  = (idx >> 4) & 31;
    int kc = (idx >> 9) & 31;
    int nT = idx >> 14;
    int K  = kc * 32 + 16 * (h >> 3) + 8 * (L >> 4) + (h & 7);
    int N  = nT * 16 + (L & 15);
    Tb[idx] = (_Float16)__expf(den_logT[K * SDEN + N]);
  }
}

struct DenShared {
  _Float16 pA[32 * 32 * 16];   // 32 KB: A fragments [kc][lane][half]
  int      pdfmap[SDEN];       // 4 KB
  float    mrow[16];
  float    wavePart[16][16];   // per-wave partial reductions
};
struct NumShared {
  _Float16 Tn[SNUM * SNUM];    // 32 KB: exp(num_logT[b]) row-major
  float    alphaN[SNUM];
  float    pN[SNUM];
  float    sPart[4 * SNUM];
  int      nmap[SNUM];
  float    m;
};
union ShMem { DenShared den; NumShared num; };

__global__ __launch_bounds__(512)
void lfmmi_kernel(const float* __restrict__ input,
                  const int*   __restrict__ seqlengths,
                  const float* __restrict__ den_init,
                  const float* __restrict__ den_final,
                  const int*   __restrict__ den_pdfmap,
                  const float* __restrict__ num_logT,
                  const float* __restrict__ num_init,
                  const float* __restrict__ num_final,
                  const int*   __restrict__ num_pdfmap,
                  const _Float16* __restrict__ Tb,
                  float* __restrict__ acc) {
  __shared__ ShMem sh;
  const int tid = threadIdx.x;
  const int blk = blockIdx.x;

  if (blk < 2) {
    // ------------- denominator: 16 batch rows / WG, 16 waves, WMMA f16 -----
    const int lane  = tid & 31;
    const int w     = tid >> 5;          // 16 waves; wave w owns cols 64w..64w+63
    const int r0    = blk * 16;
    const int lrow0 = 8 * (lane >> 4);   // this lane covers rows lrow0..lrow0+7
    const int r16   = lane & 15;
    const int cbase = w * 64 + r16;      // col = cbase + 16*tt, tt=0..3

    sh.den.pdfmap[tid]       = den_pdfmap[tid];
    sh.den.pdfmap[tid + 512] = den_pdfmap[tid + 512];
    __syncthreads();

    int   slr[8];
    float alpha[4][8];
    for (int j = 0; j < 8; ++j) slr[j] = seqlengths[r0 + lrow0 + j];
#pragma unroll
    for (int tt = 0; tt < 4; ++tt) {
      int c = cbase + 16 * tt;
      int pdf = sh.den.pdfmap[c];
      for (int j = 0; j < 8; ++j) {
        int row = lrow0 + j;
        long long off = ((long long)(r0 + row) * TLEN) * PP + pdf;
        alpha[tt][j] = den_init[c] + input[off];
      }
    }

    const _Float16* aB = &sh.den.pA[lane * 16];
    const _Float16* bB = Tb + (size_t)(4 * w) * 16384 + lane * 16;

    for (int t = 1; t < TLEN; ++t) {
      // ---- per-row max across 1024 states ----
      float lm[8];
      for (int j = 0; j < 8; ++j)
        lm[j] = fmaxf(fmaxf(alpha[0][j], alpha[1][j]),
                      fmaxf(alpha[2][j], alpha[3][j]));
      for (int mask = 1; mask <= 8; mask <<= 1)
        for (int j = 0; j < 8; ++j) lm[j] = fmaxf(lm[j], __shfl_xor(lm[j], mask, 32));
      if (r16 == 0)
        for (int j = 0; j < 8; ++j) sh.den.wavePart[w][lrow0 + j] = lm[j];
      __syncthreads();                                   // A
      if (tid < 16) {
        float mv = -3.4e38f;
        for (int ww = 0; ww < 16; ++ww) mv = fmaxf(mv, sh.den.wavePart[ww][tid]);
        sh.den.mrow[tid] = mv;
      }
      __syncthreads();                                   // B
      // ---- p = exp(alpha - m) -> LDS A fragments (WMMA layout) ----
#pragma unroll
      for (int tt = 0; tt < 4; ++tt) {
        int kcw = 2 * w + (tt >> 1);
        int h   = 8 * (tt & 1) + (r16 & 7);
        for (int j = 0; j < 8; ++j) {
          int row = lrow0 + j;
          float p = __expf(alpha[tt][j] - sh.den.mrow[row]);
          int Lw  = row + 16 * (r16 >> 3);
          sh.den.pA[(kcw * 32 + Lw) * 16 + h] = (_Float16)p;
        }
      }
      __syncthreads();                                   // C
      // ---- s = p @ T : 4 N-tiles per wave, K=1024 in 32 chunks ----
      v8f av0 = {}, av1 = {}, av2 = {}, av3 = {};
#pragma unroll 2
      for (int kc = 0; kc < 32; ++kc) {
        v16h a  = *(const v16h*)(aB + (size_t)kc * 512);
        v16h b0 = *(const v16h*)(bB + (size_t)kc * 512);
        v16h b1 = *(const v16h*)(bB + 16384 + (size_t)kc * 512);
        v16h b2 = *(const v16h*)(bB + 2 * 16384 + (size_t)kc * 512);
        v16h b3 = *(const v16h*)(bB + 3 * 16384 + (size_t)kc * 512);
        av0 = __builtin_amdgcn_wmma_f32_16x16x32_f16(false, a, false, b0, (short)0, av0, false, false);
        av1 = __builtin_amdgcn_wmma_f32_16x16x32_f16(false, a, false, b1, (short)0, av1, false, false);
        av2 = __builtin_amdgcn_wmma_f32_16x16x32_f16(false, a, false, b2, (short)0, av2, false, false);
        av3 = __builtin_amdgcn_wmma_f32_16x16x32_f16(false, a, false, b3, (short)0, av3, false, false);
      }
      // ---- alpha' = e_t + m + log(max(s,1e-30)), masked by seqlen ----
#pragma unroll
      for (int tt = 0; tt < 4; ++tt) {
        int c   = cbase + 16 * tt;
        int pdf = sh.den.pdfmap[c];
        for (int j = 0; j < 8; ++j) {
          int row = lrow0 + j;
          float s = (tt == 0) ? av0[j] : (tt == 1) ? av1[j] : (tt == 2) ? av2[j] : av3[j];
          long long off = ((long long)(r0 + row) * TLEN + t) * PP + pdf;
          float na = input[off] + sh.den.mrow[row] + __logf(fmaxf(s, 1e-30f));
          if (t < slr[j]) alpha[tt][j] = na;
        }
        if (t + 1 < TLEN) {
          long long off2 = ((long long)(r0 + lrow0) * TLEN + (t + 1)) * PP + pdf;
          __builtin_prefetch(&input[off2], 0, 0);
        }
      }
    }

    // ---- final: llh[row] = logsumexp(alpha + final) ; atomicAdd to den acc ----
    float fv[4][8];
#pragma unroll
    for (int tt = 0; tt < 4; ++tt) {
      float fc = den_final[cbase + 16 * tt];
      for (int j = 0; j < 8; ++j) fv[tt][j] = alpha[tt][j] + fc;
    }
    float lm[8];
    for (int j = 0; j < 8; ++j)
      lm[j] = fmaxf(fmaxf(fv[0][j], fv[1][j]), fmaxf(fv[2][j], fv[3][j]));
    for (int mask = 1; mask <= 8; mask <<= 1)
      for (int j = 0; j < 8; ++j) lm[j] = fmaxf(lm[j], __shfl_xor(lm[j], mask, 32));
    if (r16 == 0)
      for (int j = 0; j < 8; ++j) sh.den.wavePart[w][lrow0 + j] = lm[j];
    __syncthreads();
    if (tid < 16) {
      float mv = -3.4e38f;
      for (int ww = 0; ww < 16; ++ww) mv = fmaxf(mv, sh.den.wavePart[ww][tid]);
      sh.den.mrow[tid] = mv;
    }
    __syncthreads();
    float sm[8];
    for (int j = 0; j < 8; ++j) {
      float m2 = sh.den.mrow[lrow0 + j];
      sm[j] = __expf(fv[0][j] - m2) + __expf(fv[1][j] - m2) +
              __expf(fv[2][j] - m2) + __expf(fv[3][j] - m2);
    }
    for (int mask = 1; mask <= 8; mask <<= 1)
      for (int j = 0; j < 8; ++j) sm[j] += __shfl_xor(sm[j], mask, 32);
    __syncthreads();
    if (r16 == 0)
      for (int j = 0; j < 8; ++j) sh.den.wavePart[w][lrow0 + j] = sm[j];
    __syncthreads();
    if (tid < 16) {
      float tot = 0.0f;
      for (int ww = 0; ww < 16; ++ww) tot += sh.den.wavePart[ww][tid];
      atomicAdd(&acc[1], sh.den.mrow[tid] + __logf(tot));
    }
  } else {
    // ---------------- numerator: one utterance per WG, 128 states ----------
    const int b = blk - 2;
    for (int idx = tid; idx < SNUM * SNUM; idx += 512)
      sh.num.Tn[idx] = (_Float16)__expf(num_logT[(size_t)b * SNUM * SNUM + idx]);
    if (tid < SNUM) sh.num.nmap[tid] = num_pdfmap[b * SNUM + tid];
    __syncthreads();
    const int slv = seqlengths[b];
    if (tid < SNUM)
      sh.num.alphaN[tid] = num_init[tid] +
                           input[((long long)b * TLEN) * PP + sh.num.nmap[tid]];
    for (int t = 1; t < TLEN; ++t) {
      __syncthreads();
      if (tid < 32) {
        float v = fmaxf(fmaxf(sh.num.alphaN[tid],      sh.num.alphaN[tid + 32]),
                        fmaxf(sh.num.alphaN[tid + 64], sh.num.alphaN[tid + 96]));
        for (int mask = 16; mask >= 1; mask >>= 1) v = fmaxf(v, __shfl_xor(v, mask, 32));
        if (tid == 0) sh.num.m = v;
      }
      __syncthreads();
      const float mm = sh.num.m;
      if (tid < SNUM) sh.num.pN[tid] = __expf(sh.num.alphaN[tid] - mm);
      __syncthreads();
      {
        int j = tid & 127, q = tid >> 7;   // all 512 threads participate
        float s = 0.0f;
        for (int i = 0; i < 32; ++i)
          s = fmaf(sh.num.pN[q * 32 + i], (float)sh.num.Tn[(q * 32 + i) * SNUM + j], s);
        sh.num.sPart[q * SNUM + j] = s;
      }
      __syncthreads();
      if (tid < SNUM) {
        float s = sh.num.sPart[tid] + sh.num.sPart[SNUM + tid] +
                  sh.num.sPart[2 * SNUM + tid] + sh.num.sPart[3 * SNUM + tid];
        float e = input[((long long)b * TLEN + t) * PP + sh.num.nmap[tid]];
        float na = e + mm + __logf(fmaxf(s, 1e-30f));
        if (t < slv) sh.num.alphaN[tid] = na;
      }
    }
    __syncthreads();
    if (tid < SNUM) sh.num.pN[tid] = sh.num.alphaN[tid] + num_final[tid];
    __syncthreads();
    if (tid < 32) {
      float v = fmaxf(fmaxf(sh.num.pN[tid],      sh.num.pN[tid + 32]),
                      fmaxf(sh.num.pN[tid + 64], sh.num.pN[tid + 96]));
      for (int mask = 16; mask >= 1; mask >>= 1) v = fmaxf(v, __shfl_xor(v, mask, 32));
      float s = __expf(sh.num.pN[tid] - v)      + __expf(sh.num.pN[tid + 32] - v) +
                __expf(sh.num.pN[tid + 64] - v) + __expf(sh.num.pN[tid + 96] - v);
      for (int mask = 16; mask >= 1; mask >>= 1) s += __shfl_xor(s, mask, 32);
      if (tid == 0) atomicAdd(&acc[0], v + __logf(s));
    }
  }
}

__global__ void finalize_kernel(const float* __restrict__ acc, float* __restrict__ out) {
  if (threadIdx.x == 0 && blockIdx.x == 0) out[0] = -(acc[0] - 1.0f * acc[1]);
}

extern "C" void kernel_launch(void* const* d_in, const int* in_sizes, int n_in,
                              void* d_out, int out_size, void* d_ws, size_t ws_size,
                              hipStream_t stream) {
  const float* input      = (const float*)d_in[0];
  const int*   seqlengths = (const int*)  d_in[1];
  const float* den_logT   = (const float*)d_in[2];
  const float* den_init   = (const float*)d_in[3];
  const float* den_final  = (const float*)d_in[4];
  const int*   den_pdfmap = (const int*)  d_in[5];
  const float* num_logT   = (const float*)d_in[6];
  const float* num_init   = (const float*)d_in[7];
  const float* num_final  = (const float*)d_in[8];
  const int*   num_pdfmap = (const int*)  d_in[9];

  float*     acc = (float*)d_ws;
  _Float16*  Tb  = (_Float16*)((char*)d_ws + 256);

  // Pack exp(den_logT) into WMMA B-fragment layout (f16) + zero accumulators.
  prep_kernel<<<2048, 512, 0, stream>>>(den_logT, Tb, acc);
  // 2 den WGs (16 batch rows each, WMMA) + 32 num WGs (one utterance each).
  lfmmi_kernel<<<34, 512, 0, stream>>>(input, seqlengths, den_init, den_final,
                                       den_pdfmap, num_logT, num_init, num_final,
                                       num_pdfmap, Tb, acc);
  finalize_kernel<<<1, 1, 0, stream>>>(acc, (float*)d_out);
}
